// DGRU_36429912605229
// MI455X (gfx1250) — matrix operand order repaired and checked by reading, not compiled, not run on hardware
//
#include <hip/hip_runtime.h>
#include <math.h>

typedef __attribute__((ext_vector_type(16))) _Float16 v16h;
typedef __attribute__((ext_vector_type(8)))  float    v8f;

#define L_SEQ   2048
#define HID     128
#define TILE_B  16
#define THREADS 256

// ---- dynamic LDS layout (bytes) ----
#define OFF_UPACK 0        // 3 U mats, B-fragment packed f16: 3*4*8*512 halves = 98304 B
#define OFF_H16   98304    // h tile f16 [16][128]              = 4096 B
#define OFF_RH16  102400   // (r*h) tile f16 [16][128]          = 4096 B
#define OFF_HF    106496   // h tile f32 [16][128]              = 8192 B
#define OFF_HCAP  114688   // captured h_last f32 [16][128]     = 8192 B
#define OFF_GX    122880   // g   [16][128] f32                 = 8192 B
#define OFF_XZ    131072   // xz  [16][128] f32                 = 8192 B
#define OFF_XR    139264   // xr  [16][128] f32                 = 8192 B
#define OFF_XH    147456   // xh  [16][128] f32                 = 8192 B
#define OFF_SE    155648   // se  [16][8] f32                   = 512 B
#define OFF_SMW   156160   // Wf(36) bf(6) Wa(6) ba(1) f32      = 256 B
#define OFF_IDXT  156416   // per-batch capture step, int[16]
#define OFF_MCUR  156480   // per-batch mask this step, int[16]
#define OFF_NRM   156544   // per-batch norm, float[16]
#define OFF_SST   156608   // s staging double buffer [2][16][6] f32 = 768 B
#define SMEM_BYTES 157376

// CDNA5 hardware tanh (TRANS32 op). Trailing v_nop satisfies the
// 1-independent-op TRANS hazard rule since the compiler can't see the latency.
__device__ __forceinline__ float fast_tanh(float x) {
    float y;
    asm("v_tanh_f32 %0, %1\n\tv_nop" : "=v"(y) : "v"(x));
    return y;
}
// sigmoid(x) = 0.5*tanh(x/2) + 0.5  -> 1 trans + 2 VALU instead of div chain
__device__ __forceinline__ float sigm(float x) {
    return fmaf(fast_tanh(x * 0.5f), 0.5f, 0.5f);
}

__device__ __forceinline__ float dot6r(const float* __restrict__ w, const float* se6) {
    return w[6] + w[0]*se6[0] + w[1]*se6[1] + w[2]*se6[2]
                + w[3]*se6[3] + w[4]*se6[4] + w[5]*se6[5];
}

extern "C" __global__ __launch_bounds__(THREADS)
void dgru_fused_scan(const float* __restrict__ s,
                     const int* __restrict__ lens,
                     const unsigned char* __restrict__ mask,
                     const float* __restrict__ Wf, const float* __restrict__ bf,
                     const float* __restrict__ Wa, const float* __restrict__ ba,
                     const float* __restrict__ Wg, const float* __restrict__ bg,
                     const float* __restrict__ Wz, const float* __restrict__ bz,
                     const float* __restrict__ Wr, const float* __restrict__ br,
                     const float* __restrict__ Wh, const float* __restrict__ bh,
                     const float* __restrict__ Uz, const float* __restrict__ Ur,
                     const float* __restrict__ Uh,
                     float* __restrict__ out)
{
    extern __shared__ char smem[];
    _Float16* upack  = (_Float16*)(smem + OFF_UPACK);
    _Float16* h16    = (_Float16*)(smem + OFF_H16);
    _Float16* rh16   = (_Float16*)(smem + OFF_RH16);
    float*    hf     = (float*)(smem + OFF_HF);
    float*    hcap   = (float*)(smem + OFF_HCAP);
    float*    gx     = (float*)(smem + OFF_GX);
    float*    xz     = (float*)(smem + OFF_XZ);
    float*    xr     = (float*)(smem + OFF_XR);
    float*    xh     = (float*)(smem + OFF_XH);
    float*    sebuf  = (float*)(smem + OFF_SE);
    float*    smw    = (float*)(smem + OFF_SMW);
    int*      idxt   = (int*)(smem + OFF_IDXT);
    int*      mcur   = (int*)(smem + OFF_MCUR);
    float*    nrm    = (float*)(smem + OFF_NRM);
    float*    sstage = (float*)(smem + OFF_SST);   // [2][96]

    const int tid  = threadIdx.x;
    const int lane = tid & 31;
    const int wave = tid >> 5;          // n-tile owner: n in [16*wave, 16*wave+16)
    const int b0   = blockIdx.x * TILE_B;

    // ---------------- one-time init ----------------
    // Pack Uz/Ur/Uh into WMMA B-fragment order (f16). B[k][n] = U[n][k].
    for (int p = tid; p < 3 * 4 * 8 * 512; p += THREADS) {
        const int e  = p & 15;
        const int ln = (p >> 4) & 31;
        const int nt = (p >> 9) & 7;
        const int kb = (p >> 12) & 3;
        const int u  = p >> 14;
        const int N  = nt * 16 + (ln & 15);
        const int K  = kb * 32 + ((e >> 3) << 4) + ((ln >> 4) << 3) + (e & 7);
        const float* Uu = (u == 0) ? Uz : ((u == 1) ? Ur : Uh);
        upack[p] = (_Float16)Uu[N * HID + K];
    }
    for (int p = tid; p < TILE_B * HID; p += THREADS) {
        hf[p] = 0.0f; hcap[p] = 0.0f;
        h16[p] = (_Float16)0.0f; rh16[p] = (_Float16)0.0f;
    }
    if (tid < 36)            smw[tid] = Wf[tid];
    else if (tid < 42)       smw[tid] = bf[tid - 36];
    else if (tid < 48)       smw[tid] = Wa[tid - 42];
    else if (tid == 48)      smw[48]  = ba[0];
    if (tid < TILE_B) {
        int l = lens[b0 + tid];
        idxt[tid] = (l < 1 ? 1 : l) - 1;
    }
    // Per-thread channel weights in registers: each thread owns channel n=tid&127
    const int nch = tid & 127;
    float wgr[7], wzr[7], wrr[7], whr[7];
#pragma unroll
    for (int j = 0; j < 6; ++j) {
        wgr[j] = Wg[nch * 6 + j]; wzr[j] = Wz[nch * 6 + j];
        wrr[j] = Wr[nch * 6 + j]; whr[j] = Wh[nch * 6 + j];
    }
    wgr[6] = bg[nch]; wzr[6] = bz[nch]; wrr[6] = br[nch]; whr[6] = bh[nch];

    // Async double-buffered staging of s tiles (96 dwords / step).
    // LDS generic address low 32 bits == LDS byte offset (shared aperture).
    const unsigned ldsS0 = (unsigned)(uintptr_t)(sstage + tid);
    const unsigned ldsS1 = (unsigned)(uintptr_t)(sstage + 96 + tid);
    const float* sgbase = s;
    if (tid < 96) {
        const int srow = tid / 6, sc = tid - srow * 6;
        sgbase = s + (size_t)(b0 + srow) * L_SEQ * 6 + sc;
        // kick off tile for t=0 into buffer 0
        asm volatile("global_load_async_to_lds_b32 %0, %1, off"
                     :: "v"(ldsS0), "v"(sgbase) : "memory");
        asm volatile("s_wait_asynccnt 0x0" ::: "memory");
    }
    __syncthreads();

    // Fragment index helpers (constant across steps)
    const int ma    = lane & 15;               // A-fragment row (batch in tile)
    const int khalf = (lane >> 4) << 3;        // 0 or 8 (A-fragment K sub-offset)
    const int mb    = (lane >> 4) << 3;        // C/D fragment: M = r + mb
    const int nloc  = wave * 16 + (lane & 15); // C/D fragment: column

    // ---------------- sequential scan ----------------
    for (int t = 0; t < L_SEQ; ++t) {
        // (0) kick off async fetch of next step's s tile (overlaps the GEMMs)
        if (t + 1 < L_SEQ && tid < 96) {
            const float* ga = sgbase + (size_t)(t + 1) * 6;
            asm volatile("global_load_async_to_lds_b32 %0, %1, off"
                         :: "v"(((t + 1) & 1) ? ldsS1 : ldsS0), "v"(ga) : "memory");
        }

        // (a) front-end part 1: se + mask (one thread per batch row)
        if (tid < TILE_B) {
            const float* sp = sstage + (t & 1) * 96 + tid * 6;
            float sv[6], fv[6];
#pragma unroll
            for (int i = 0; i < 6; ++i) sv[i] = sp[i];
#pragma unroll
            for (int j = 0; j < 6; ++j) {
                float acc = smw[36 + j];
#pragma unroll
                for (int i = 0; i < 6; ++i) acc += smw[j * 6 + i] * sv[i];
                fv[j] = acc;
            }
            float aacc = smw[48];
#pragma unroll
            for (int j = 0; j < 6; ++j) aacc += smw[42 + j] * fv[j];
            const float al = sigm(aacc);
#pragma unroll
            for (int j = 0; j < 6; ++j) sebuf[tid * 8 + j] = sv[j] + al * fv[j];
            mcur[tid] = (int)mask[(size_t)(b0 + tid) * L_SEQ + t];
        }
        __syncthreads();

        // (b) front-end part 2: g / xz / xr / xh projections (register weights)
#pragma unroll
        for (int i = 0; i < 8; ++i) {
            const int p = tid + THREADS * i;     // (m = (tid>>7)+2i, n = tid&127)
            const float* se6 = sebuf + (p >> 7) * 8;
            gx[p] = sigm(dot6r(wgr, se6));
            xz[p] = dot6r(wzr, se6);
            xr[p] = dot6r(wrr, se6);
            xh[p] = dot6r(whr, se6);
        }
        __syncthreads();

        // (c) GEMM1: accz = h@Uz.T, accr = h@Ur.T for this wave's n-tile
        v8f accz = {}; v8f accr = {};
#pragma unroll
        for (int kb = 0; kb < 4; ++kb) {
            union { v16h v; uint4 q[2]; } afr, bzf, brf;
            const _Float16* ap = h16 + ma * HID + kb * 32 + khalf;
            afr.q[0] = *(const uint4*)(ap);
            afr.q[1] = *(const uint4*)(ap + 16);
            const _Float16* bzp = upack + (((0 * 4 + kb) * 8 + wave) << 9) + (lane << 4);
            bzf.q[0] = ((const uint4*)bzp)[0];
            bzf.q[1] = ((const uint4*)bzp)[1];
            const _Float16* brp = upack + (((1 * 4 + kb) * 8 + wave) << 9) + (lane << 4);
            brf.q[0] = ((const uint4*)brp)[0];
            brf.q[1] = ((const uint4*)brp)[1];
            accz = __builtin_amdgcn_wmma_f32_16x16x32_f16(false, afr.v, false, bzf.v,
                                                          (short)0, accz, false, false);
            accr = __builtin_amdgcn_wmma_f32_16x16x32_f16(false, afr.v, false, brf.v,
                                                          (short)0, accr, false, false);
        }

        // (d) elementwise 1: z (kept in regs), r, stage (r*h) as f16 for GEMM2
        float zf[8];
#pragma unroll
        for (int r = 0; r < 8; ++r) {
            const int p = (r + mb) * HID + nloc;
            zf[r] = sigm(xz[p] + accz[r]);
            const float rr = sigm(xr[p] + accr[r]);
            rh16[p] = (_Float16)(rr * hf[p]);
        }
        __syncthreads();

        // (e) GEMM2: acch = (r*h)@Uh.T
        v8f acch = {};
#pragma unroll
        for (int kb = 0; kb < 4; ++kb) {
            union { v16h v; uint4 q[2]; } afr, bhf;
            const _Float16* ap = rh16 + ma * HID + kb * 32 + khalf;
            afr.q[0] = *(const uint4*)(ap);
            afr.q[1] = *(const uint4*)(ap + 16);
            const _Float16* bhp = upack + (((2 * 4 + kb) * 8 + wave) << 9) + (lane << 4);
            bhf.q[0] = ((const uint4*)bhp)[0];
            bhf.q[1] = ((const uint4*)bhp)[1];
            acch = __builtin_amdgcn_wmma_f32_16x16x32_f16(false, afr.v, false, bhf.v,
                                                          (short)0, acch, false, false);
        }

        // (f) elementwise 2: h_tilde, gate blend, mask, capture
#pragma unroll
        for (int r = 0; r < 8; ++r) {
            const int m = r + mb;
            const int p = m * HID + nloc;
            const float ht   = fast_tanh(xh[p] + acch[r]);
            const float z    = zf[r];
            const float hold = hf[p];
            const float hn   = (1.0f - z) * hold + z * (gx[p] * ht);
            const float hout = mcur[m] ? hold : hn;
            hf[p]  = hout;
            h16[p] = (_Float16)hout;
            if (t == idxt[m]) hcap[p] = hout;
        }

        // (g) drain the async fetch issued for t+1, then publish step state
        if (t + 1 < L_SEQ && tid < 96) {
            asm volatile("s_wait_asynccnt 0x0" ::: "memory");
        }
        __syncthreads();
    }

    // ---------------- finalize: L2 normalize captured states ----------------
    if (tid < TILE_B) {
        float ss = 0.0f;
        for (int n = 0; n < HID; ++n) { float v = hcap[tid * HID + n]; ss += v * v; }
        float nr = sqrtf(ss);
        nrm[tid] = (nr > 1e-12f) ? nr : 1e-12f;
    }
    __syncthreads();
#pragma unroll
    for (int i = 0; i < 8; ++i) {
        const int p = tid + THREADS * i;
        const int m = p >> 7;
        const int n = p & 127;
        out[(size_t)(b0 + m) * HID + n] = hcap[p] / nrm[m];
    }
}

extern "C" void kernel_launch(void* const* d_in, const int* in_sizes, int n_in,
                              void* d_out, int out_size, void* d_ws, size_t ws_size,
                              hipStream_t stream) {
    (void)in_sizes; (void)n_in; (void)d_ws; (void)ws_size; (void)out_size;
    const float*         s    = (const float*)d_in[0];
    const int*           lens = (const int*)d_in[1];
    const unsigned char* mask = (const unsigned char*)d_in[2];
    const float* Wf = (const float*)d_in[3];  const float* bf = (const float*)d_in[4];
    const float* Wa = (const float*)d_in[5];  const float* ba = (const float*)d_in[6];
    const float* Wg = (const float*)d_in[7];  const float* bg = (const float*)d_in[8];
    const float* Wz = (const float*)d_in[9];  const float* bz = (const float*)d_in[10];
    const float* Wr = (const float*)d_in[11]; const float* br = (const float*)d_in[12];
    const float* Wh = (const float*)d_in[13]; const float* bh = (const float*)d_in[14];
    const float* Uz = (const float*)d_in[15];
    const float* Ur = (const float*)d_in[16];
    const float* Uh = (const float*)d_in[17];
    float* out = (float*)d_out;

    (void)hipFuncSetAttribute(reinterpret_cast<const void*>(dgru_fused_scan),
                              hipFuncAttributeMaxDynamicSharedMemorySize, SMEM_BYTES);
    dgru_fused_scan<<<dim3(256 / TILE_B), dim3(THREADS), SMEM_BYTES, stream>>>(
        s, lens, mask, Wf, bf, Wa, ba, Wg, bg, Wz, bz, Wr, br, Wh, bh, Uz, Ur, Uh, out);
}
